// SetAbstractionMSG_24781961298015
// MI455X (gfx1250) — matrix operand order repaired and checked by reading, not compiled, not run on hardware
//
#include <hip/hip_runtime.h>

typedef __attribute__((ext_vector_type(16))) _Float16 v16h;
typedef __attribute__((ext_vector_type(8)))  _Float16 v8h;
typedef __attribute__((ext_vector_type(8)))  float    v8f;

static constexpr int BATCH = 8;
static constexpr int NPTS  = 4096;
static constexpr int SCENT = 1024;
static constexpr int INCH  = 64;

struct LayerP { const float *w, *b, *gamma, *beta, *mean, *var; };

// Per-branch device-side prepared data: swizzled f16 weights + folded BN.
struct DevBranch {
  const _Float16* swz[3];   // B-fragment-layout weights, K zero-padded
  const float*    sb[3];    // [0,Cout): scale, [Cout,2*Cout): bias
};

// ---------------------------------------------------------------------------
// Kernel 0: weight prep. Swizzle w (Cout x Cin, f32) into a per-lane (contiguous-k) WMMA
// B-fragment layout as f16 with K padded to KS*32, and fold BN into per-channel
// scale/bias. Layout: swz[((nT*KS + ks)*32 + lane)*16 + e], where
//   n = nT*16 + (lane&15), k = ks*32 + (lane>>4)*16 + e.
// => in the MLP hot loop a lane's whole B fragment is one aligned 32B load.
// ---------------------------------------------------------------------------
__global__ void __launch_bounds__(256) prep_kernel(LayerP lp, int Cin, int Cout,
                                                   int KS, _Float16* __restrict__ swz,
                                                   float* __restrict__ sb) {
  const int i = blockIdx.x * 256 + threadIdx.x;
  const int total = Cout * KS * 32;  // total f16 elements in swizzled blob
  if (i < total) {
    const int e  = i & 15;
    int rest     = i >> 4;
    const int L  = rest & 31; rest >>= 5;
    const int ks = rest % KS;
    const int nT = rest / KS;
    const int n  = nT * 16 + (L & 15);
    const int k  = ks * 32 + (L >> 4) * 16 + e;
    swz[i] = (k < Cin) ? (_Float16)lp.w[n * Cin + k] : (_Float16)0.0f;
  }
  if (i < Cout) {
    const float inv = lp.gamma[i] * rsqrtf(lp.var[i] + 1e-5f);
    sb[i]        = inv;
    sb[Cout + i] = lp.b[i] * inv + lp.beta[i] - lp.mean[i] * inv;
  }
}

// ---------------------------------------------------------------------------
// Kernel 1: Farthest Point Sampling. One block per batch; LDS-resident
// min-dist array; 1024 serial (update, block-argmax) iterations.
// ---------------------------------------------------------------------------
__global__ void __launch_bounds__(256) fps_kernel(const float* __restrict__ xyz,
                                                  float* __restrict__ out_cent,
                                                  float* __restrict__ ws_cent,
                                                  int*   __restrict__ ws_idx) {
  __shared__ float sdist[NPTS];
  __shared__ float rd[256];
  __shared__ int   ri[256];
  const int b = blockIdx.x, tid = threadIdx.x;
  const float* px = xyz + (size_t)b * NPTS * 3;
  for (int p = tid; p < NPTS; p += 256) sdist[p] = 1e10f;
  __syncthreads();
  int far = 0;
  for (int it = 0; it < SCENT; ++it) {
    const float cx = px[far * 3 + 0];
    const float cy = px[far * 3 + 1];
    const float cz = px[far * 3 + 2];
    if (tid == 0) {
      ws_idx[b * SCENT + it] = far;
      const size_t o = ((size_t)b * SCENT + it) * 3;
      out_cent[o + 0] = cx; out_cent[o + 1] = cy; out_cent[o + 2] = cz;
      ws_cent[o + 0]  = cx; ws_cent[o + 1]  = cy; ws_cent[o + 2]  = cz;
    }
    float bd = -1.0f; int bi = 0;
    for (int p = tid; p < NPTS; p += 256) {
      const float dx = px[p * 3 + 0] - cx;
      const float dy = px[p * 3 + 1] - cy;
      const float dz = px[p * 3 + 2] - cz;
      const float d  = dx * dx + dy * dy + dz * dz;
      const float nd = fminf(sdist[p], d);
      sdist[p] = nd;
      if (nd > bd) { bd = nd; bi = p; }
    }
    rd[tid] = bd; ri[tid] = bi;
    __syncthreads();
    for (int s = 128; s > 0; s >>= 1) {
      if (tid < s) {
        const float od = rd[tid + s]; const int oi = ri[tid + s];
        if (od > rd[tid] || (od == rd[tid] && oi < ri[tid])) { rd[tid] = od; ri[tid] = oi; }
      }
      __syncthreads();
    }
    far = ri[0];
    __syncthreads();
  }
}

// ---------------------------------------------------------------------------
// Kernel 2: Ball query (3 radii fused). First k in-radius indices in ascending
// order (== reference sort-then-slice), filled with the first hit.
// ---------------------------------------------------------------------------
__global__ void __launch_bounds__(128) ballq_kernel(const float* __restrict__ xyz,
                                                    const float* __restrict__ cent,
                                                    int* __restrict__ g1,
                                                    int* __restrict__ g2,
                                                    int* __restrict__ g3) {
  const int bs = blockIdx.x * 128 + threadIdx.x;
  if (bs >= BATCH * SCENT) return;
  const int b = bs >> 10;
  const float* px = xyz + (size_t)b * NPTS * 3;
  const float cx = cent[bs * 3 + 0], cy = cent[bs * 3 + 1], cz = cent[bs * 3 + 2];
  int c1 = 0, c2 = 0, c3 = 0;
  for (int p = 0; p < NPTS; ++p) {
    if (c1 >= 16 && c2 >= 32 && c3 >= 64) break;
    const float dx = px[p * 3 + 0] - cx;
    const float dy = px[p * 3 + 1] - cy;
    const float dz = px[p * 3 + 2] - cz;
    const float d  = dx * dx + dy * dy + dz * dz;
    if (d <= 0.04f && c1 < 16) g1[bs * 16 + c1++] = p;
    if (d <= 0.16f && c2 < 32) g2[bs * 32 + c2++] = p;
    if (d <= 0.64f && c3 < 64) g3[bs * 64 + c3++] = p;
  }
  int f;
  f = c1 ? g1[bs * 16] : (NPTS - 1); for (; c1 < 16; ++c1) g1[bs * 16 + c1] = f;
  f = c2 ? g2[bs * 32] : (NPTS - 1); for (; c2 < 32; ++c2) g2[bs * 32 + c2] = f;
  f = c3 ? g3[bs * 64] : (NPTS - 1); for (; c3 < 64; ++c3) g3[bs * 64 + c3] = f;
}

// ---------------------------------------------------------------------------
// WMMA layer: D = relu((A x W^T) * scale + bias).
//   A (M x Cin f16, LDS row-major): lane's fragment = two contiguous v8h LDS
//     loads (ds_load_b128) per K-step.
//   B: one aligned v16h load from the pre-swizzled global blob (2x b128).
//   C/D: n = lane&15, m = r + 8*(lane>>4).
// Tile loop is wave-uniform so EXEC is all ones at every v_wmma.
// ---------------------------------------------------------------------------
template <int M, int Win, int Cout, int KS, int Wout>
__device__ inline void mlp_layer(const _Float16* __restrict__ in,
                                 _Float16* __restrict__ out,
                                 const _Float16* __restrict__ swz,
                                 const float* __restrict__ sb, int tid) {
  constexpr int MT = M / 16, NT = Cout / 16, TILES = MT * NT;
  const int wave = tid >> 5, lane = tid & 31;
  const int n16 = lane & 15, hlf = lane >> 4;
  const v16h* __restrict__ wfrag = (const v16h*)swz;
  for (int t = wave; t < TILES; t += 4) {
    const int mT = t / NT, nT = t % NT;
    const int nG = nT * 16 + n16;
    const float scale = sb[nG];
    const float bias  = sb[Cout + nG];
    const int mrow = mT * 16 + n16;
    v8f acc = {};
#pragma unroll
    for (int ks = 0; ks < KS; ++ks) {
      const v8h alo = *(const v8h*)(in + mrow * Win + ks * 32 + hlf * 16);
      const v8h ahi = *(const v8h*)(in + mrow * Win + ks * 32 + hlf * 16 + 8);
      const v16h av = __builtin_shufflevector(alo, ahi, 0, 1, 2, 3, 4, 5, 6, 7,
                                              8, 9, 10, 11, 12, 13, 14, 15);
      const v16h bv = wfrag[(nT * KS + ks) * 32 + lane];
      acc = __builtin_amdgcn_wmma_f32_16x16x32_f16(false, av, false, bv,
                                                   (short)0, acc, false, false);
    }
#pragma unroll
    for (int r = 0; r < 8; ++r) {
      float v = acc[r] * scale + bias;
      v = v > 0.0f ? v : 0.0f;
      out[(mT * 16 + r + hlf * 8) * Wout + nG] = (_Float16)v;
    }
  }
}

// ---------------------------------------------------------------------------
// Kernel 3 (per branch): gather -> 3x (WMMA + fused BN + ReLU) -> max over k.
// 4 waves per centroid; f16 activations ping-pong in LDS.
// ---------------------------------------------------------------------------
template <int KSAMP, int C1, int C2, int C3>
__global__ void __launch_bounds__(128) samsg_mlp_kernel(
    const float* __restrict__ xyz, const float* __restrict__ feats,
    const float* __restrict__ cent, const int* __restrict__ gidx,
    DevBranch db, float* __restrict__ out_feat, int chOff, int wbytes) {
  constexpr int WA = (96 > C2) ? 96 : C2;
  constexpr int WB = (C1 > C3) ? C1 : C3;
  __shared__ alignas(32) _Float16 bufA[KSAMP * WA];
  __shared__ alignas(32) _Float16 bufB[KSAMP * WB];
  const int bs = blockIdx.x, tid = threadIdx.x;
  const int b = bs >> 10;

  // Warm L2/L0 with this branch's swizzled weights (global_prefetch_b8).
  {
    const char* pf = (const char*)db.swz[0];
    for (int o = tid * 256; o < wbytes; o += 128 * 256) __builtin_prefetch(pf + o, 0, 1);
  }

  // Gather: channels [0,64)=feats, [64,67)=xyz - centroid, [67,96)=zero pad.
  for (int e = tid; e < KSAMP * 96; e += 128) {
    const int j = e / 96, c = e % 96;
    const int idx = gidx[bs * KSAMP + j];
    float v = 0.0f;
    if (c < 64) {
      v = feats[((size_t)b * NPTS + idx) * INCH + c];
    } else if (c < 67) {
      const int d = c - 64;
      v = xyz[((size_t)b * NPTS + idx) * 3 + d] - cent[bs * 3 + d];
    }
    bufA[j * WA + c] = (_Float16)v;
  }
  __syncthreads();
  mlp_layer<KSAMP, WA, C1, 3, WB>(bufA, bufB, db.swz[0], db.sb[0], tid);
  __syncthreads();
  mlp_layer<KSAMP, WB, C2, C1 / 32, WA>(bufB, bufA, db.swz[1], db.sb[1], tid);
  __syncthreads();
  mlp_layer<KSAMP, WA, C3, C2 / 32, WB>(bufA, bufB, db.swz[2], db.sb[2], tid);
  __syncthreads();

  for (int c = tid; c < C3; c += 128) {
    float m = -1e30f;
    for (int j = 0; j < KSAMP; ++j) m = fmaxf(m, (float)bufB[j * WB + c]);
    out_feat[(size_t)bs * 640 + chOff + c] = m;
  }
}

// ---------------------------------------------------------------------------
extern "C" void kernel_launch(void* const* d_in, const int* in_sizes, int n_in,
                              void* d_out, int out_size, void* d_ws, size_t ws_size,
                              hipStream_t stream) {
  const float* xyz   = (const float*)d_in[0];
  const float* feats = (const float*)d_in[1];

  // params pytree leaves: per layer, dict keys alphabetical: b,beta,gamma,mean,var,w
  LayerP lp[3][3];
  for (int br = 0; br < 3; ++br) {
    for (int ly = 0; ly < 3; ++ly) {
      const int base = 2 + (br * 3 + ly) * 6;
      lp[br][ly].b     = (const float*)d_in[base + 0];
      lp[br][ly].beta  = (const float*)d_in[base + 1];
      lp[br][ly].gamma = (const float*)d_in[base + 2];
      lp[br][ly].mean  = (const float*)d_in[base + 3];
      lp[br][ly].var   = (const float*)d_in[base + 4];
      lp[br][ly].w     = (const float*)d_in[base + 5];
    }
  }

  static const int CIN[3][3]  = {{67, 64, 64}, {67, 128, 128}, {67, 128, 128}};
  static const int COUT[3][3] = {{64, 64, 128}, {128, 128, 256}, {128, 128, 256}};

  auto align64 = [](size_t x) { return (x + 63) & ~(size_t)63; };
  char* base = (char*)d_ws;
  size_t off = 0;
  float* ws_cent = (float*)(base + off); off = align64(off + (size_t)BATCH * SCENT * 3 * sizeof(float));
  int*   ws_idx  = (int*)(base + off);   off = align64(off + (size_t)BATCH * SCENT * sizeof(int));
  int*   g1      = (int*)(base + off);   off = align64(off + (size_t)BATCH * SCENT * 16 * sizeof(int));
  int*   g2      = (int*)(base + off);   off = align64(off + (size_t)BATCH * SCENT * 32 * sizeof(int));
  int*   g3      = (int*)(base + off);   off = align64(off + (size_t)BATCH * SCENT * 64 * sizeof(int));

  DevBranch db[3];
  int wbytes[3];
  _Float16* swz[3][3];
  float*    sb[3][3];
  for (int br = 0; br < 3; ++br) {
    int total_bytes = 0;
    for (int ly = 0; ly < 3; ++ly) {
      const int KS = (CIN[br][ly] + 31) / 32;
      const int nswz = COUT[br][ly] * KS * 32;  // f16 elements
      swz[br][ly] = (_Float16*)(base + off); off = align64(off + (size_t)nswz * 2);
      sb[br][ly]  = (float*)(base + off);    off = align64(off + (size_t)2 * COUT[br][ly] * 4);
      db[br].swz[ly] = swz[br][ly];
      db[br].sb[ly]  = sb[br][ly];
      total_bytes += nswz * 2;
    }
    wbytes[br] = total_bytes;  // swz blobs are adjacent-ish; prefetch span
  }

  // Weight prep (9 tiny launches).
  for (int br = 0; br < 3; ++br) {
    for (int ly = 0; ly < 3; ++ly) {
      const int KS = (CIN[br][ly] + 31) / 32;
      const int total = COUT[br][ly] * KS * 32;
      prep_kernel<<<(total + 255) / 256, 256, 0, stream>>>(
          lp[br][ly], CIN[br][ly], COUT[br][ly], KS, swz[br][ly], sb[br][ly]);
    }
  }

  float* out      = (float*)d_out;
  float* out_feat = out + (size_t)BATCH * SCENT * 3;

  fps_kernel<<<BATCH, 256, 0, stream>>>(xyz, out, ws_cent, ws_idx);
  ballq_kernel<<<(BATCH * SCENT + 127) / 128, 128, 0, stream>>>(xyz, ws_cent, g1, g2, g3);
  samsg_mlp_kernel<16, 64, 64, 128><<<BATCH * SCENT, 128, 0, stream>>>(
      xyz, feats, ws_cent, g1, db[0], out_feat, 0, wbytes[0]);
  samsg_mlp_kernel<32, 128, 128, 256><<<BATCH * SCENT, 128, 0, stream>>>(
      xyz, feats, ws_cent, g2, db[1], out_feat, 128, wbytes[1]);
  samsg_mlp_kernel<64, 128, 128, 256><<<BATCH * SCENT, 128, 0, stream>>>(
      xyz, feats, ws_cent, g3, db[2], out_feat, 384, wbytes[2]);
}
